// WindowedTemporalAttention_62491774157534
// MI455X (gfx1250) — compile-verified, matrix operand order checked by reading, not run on hardware
//
#include <hip/hip_runtime.h>

typedef __attribute__((ext_vector_type(16))) _Float16 v16h;
typedef __attribute__((ext_vector_type(8)))  _Float16 v8h;
typedef __attribute__((ext_vector_type(8)))  float    v8f;

constexpr int NBATCH = 2;
constexpr int TT     = 1024;
constexpr int DD     = 1024;
constexpr int HH     = 16;
constexpr int DHH    = 64;
constexpr int WR     = 64;     // half window
constexpr int SPAN   = 144;    // 16 + 2*WR
constexpr int SPANP  = 160;    // padded to multiple of 32

// ---------------------------------------------------------------------------
// Fragment loader matching CDNA5 16-bit A/B VGPR layout (ISA 7.12.2):
// lane l holds K = {kb..kb+7, kb+16..kb+23}, kb = (l>=16) ? 8 : 0.
// p points at the start of this lane's 32-long K segment (row for A, col for B).
// ---------------------------------------------------------------------------
__device__ __forceinline__ v16h load_frag(const _Float16* p, int lane) {
    const int kb = (lane & 16) ? 8 : 0;
    v8h lo = *(const v8h*)(p + kb);
    v8h hi = *(const v8h*)(p + kb + 16);
    v16h r;
#pragma unroll
    for (int i = 0; i < 8; ++i) { r[i] = lo[i]; r[i + 8] = hi[i]; }
    return r;
}

__device__ __forceinline__ v8f wmma16(v16h a, v16h b, v8f c) {
    return __builtin_amdgcn_wmma_f32_16x16x32_f16(
        /*neg_a=*/false, a, /*neg_b=*/false, b,
        /*c_mod=*/(short)0, c, /*reuse_a=*/false, /*reuse_b=*/false);
}

// LDS byte offset of a generic pointer to __shared__ (low 32 bits of flat addr)
__device__ __forceinline__ uint32_t lds_off(const void* p) {
    return (uint32_t)(uintptr_t)p;
}

// Async global->LDS 16B copy per lane (gfx1250, tracked by ASYNCcnt)
#define ASYNC_CP_B128(ldsaddr, gptr)                                           \
    asm volatile("global_load_async_to_lds_b128 %0, %1, off"                   \
                 :: "v"(ldsaddr), "v"(gptr) : "memory")

// ---------------------------------------------------------------------------
// fp32 -> fp16 cast (one-time, so GEMMs re-read weights as fp16 from L2)
// ---------------------------------------------------------------------------
__global__ void cast_f32_to_f16(const float* __restrict__ in,
                                _Float16* __restrict__ out, int n) {
    for (int i = blockIdx.x * blockDim.x + threadIdx.x; i < n;
         i += gridDim.x * blockDim.x)
        out[i] = (_Float16)in[i];
}

// ---------------------------------------------------------------------------
// GEMM core: out(MxN) = A(MxK) @ W^T + b, M rows of A row-major, W row-major
// (so both contract along contiguous K).  Block: 256 thr = 8 waves.
// Block tile: 256(M) x 64(N); wave tile: 32(M) x 64(N).
// Weight tile 64x32 f16 (4 KB) staged in LDS via async-to-LDS, double buffered,
// shared by all 8 waves. A streams from global (own rows, b128 pairs).
// ---------------------------------------------------------------------------
template <bool F16OUT>
__device__ __forceinline__ void gemm_core(
    const _Float16* __restrict__ a16, const _Float16* __restrict__ w16,
    const float* __restrict__ bias, void* __restrict__ outp,
    _Float16* __restrict__ Bl0, _Float16* __restrict__ Bl1)
{
    const int tid  = threadIdx.x;
    const int lane = tid & 31;
    const int wave = tid >> 5;
    const int ln = lane & 15;
    const int hb = (lane >> 4) << 3;
    const int m0 = (blockIdx.y * 8 + wave) * 32;
    const int n0 = blockIdx.x * 64;

    // cooperative async-copy assignment: 64 rows x 32 halves = 256 x 16B
    const int crow = tid >> 2;            // 0..63
    const int cchk = (tid & 3) * 8;       // half offset 0,8,16,24
    const _Float16* gsrc = w16 + (size_t)(n0 + crow) * DD + cchk;
    const uint32_t l0 = lds_off(Bl0 + crow * 32 + cchk);
    const uint32_t l1 = lds_off(Bl1 + crow * 32 + cchk);

    ASYNC_CP_B128(l0, gsrc);              // stage k0 = 0 into buffer 0

    v8f acc[2][4] = {};
    const _Float16* arow0 = a16 + (size_t)(m0 + ln) * DD;
    const _Float16* arow1 = a16 + (size_t)(m0 + 16 + ln) * DD;

    constexpr int NK = DD / 32;
#pragma unroll 2
    for (int ki = 0; ki < NK; ++ki) {
        const int k0 = ki * 32;
        if (ki + 1 < NK) {
            ASYNC_CP_B128((ki & 1) ? l0 : l1, gsrc + k0 + 32);
            asm volatile("s_wait_asynccnt 0x1" ::: "memory");
        } else {
            asm volatile("s_wait_asynccnt 0x0" ::: "memory");
        }
        __syncthreads();                  // whole 64x32 tile visible

        __builtin_prefetch(arow0 + k0 + 256, 0, 3);
        const _Float16* bb = (ki & 1) ? Bl1 : Bl0;
        v16h a0 = load_frag(arow0 + k0, lane);
        v16h a1 = load_frag(arow1 + k0, lane);
        v16h b0 = load_frag(bb + (0 * 16 + ln) * 32, lane);
        v16h b1 = load_frag(bb + (1 * 16 + ln) * 32, lane);
        v16h b2 = load_frag(bb + (2 * 16 + ln) * 32, lane);
        v16h b3 = load_frag(bb + (3 * 16 + ln) * 32, lane);

        acc[0][0] = wmma16(a0, b0, acc[0][0]);
        acc[0][1] = wmma16(a0, b1, acc[0][1]);
        acc[0][2] = wmma16(a0, b2, acc[0][2]);
        acc[0][3] = wmma16(a0, b3, acc[0][3]);
        acc[1][0] = wmma16(a1, b0, acc[1][0]);
        acc[1][1] = wmma16(a1, b1, acc[1][1]);
        acc[1][2] = wmma16(a1, b2, acc[1][2]);
        acc[1][3] = wmma16(a1, b3, acc[1][3]);

        __syncthreads();                  // done reading before next overwrite
    }

#pragma unroll
    for (int mt = 0; mt < 2; ++mt)
#pragma unroll
        for (int nt = 0; nt < 4; ++nt) {
            const int n = n0 + nt * 16 + ln;
            const float bv = bias[n];
#pragma unroll
            for (int r = 0; r < 8; ++r) {
                const size_t idx = (size_t)(m0 + mt * 16 + hb + r) * DD + n;
                if (F16OUT)
                    ((_Float16*)outp)[idx] = (_Float16)(acc[mt][nt][r] + bv);
                else
                    ((float*)outp)[idx] = acc[mt][nt][r] + bv;
            }
        }
}

__global__ __launch_bounds__(256) void gemm_qkv(
    const _Float16* __restrict__ x16,
    const _Float16* __restrict__ wq16, const _Float16* __restrict__ wk16,
    const _Float16* __restrict__ wv16,
    const float* __restrict__ bq, const float* __restrict__ bk,
    const float* __restrict__ bv,
    _Float16* __restrict__ qo, _Float16* __restrict__ ko,
    _Float16* __restrict__ vo)
{
    __shared__ _Float16 Bl[2][64 * 32];   // 2 x 4 KB double buffer
    const _Float16* w16;
    const float*    bias;
    _Float16*       out;
    if (blockIdx.z == 0)      { w16 = wq16; bias = bq; out = qo; }
    else if (blockIdx.z == 1) { w16 = wk16; bias = bk; out = ko; }
    else                      { w16 = wv16; bias = bv; out = vo; }
    gemm_core<true>(x16, w16, bias, out, Bl[0], Bl[1]);
}

__global__ __launch_bounds__(256) void gemm_out(
    const _Float16* __restrict__ a16, const _Float16* __restrict__ wo16,
    const float* __restrict__ bo, float* __restrict__ out)
{
    __shared__ _Float16 Bl[2][64 * 32];
    gemm_core<false>(a16, wo16, bo, out, Bl[0], Bl[1]);
}

// ---------------------------------------------------------------------------
// Fused windowed attention. One wave per (batch, head, 16-query tile).
// Key span for the tile = [t0-64, t0+79] (144), padded to 160.
// S = Q Kspan^T in 9 accumulator tiles; mask+softmax across 16-lane halves
// via __shfl_xor(..,16); P staged to LDS; V span transposed into LDS so the
// PV B-fragments are contiguous; O = P Vspan via 20 WMMAs.
// ---------------------------------------------------------------------------
__global__ __launch_bounds__(32) void attn_win(
    const _Float16* __restrict__ q, const _Float16* __restrict__ k,
    const _Float16* __restrict__ v, _Float16* __restrict__ o)
{
    __shared__ _Float16 Pb[16 * SPANP];        // 5.0 KB
    __shared__ _Float16 Vt[DHH * SPANP];       // 20.0 KB  (Vt[d][j] = v[pos(j), d])

    const int lane = threadIdx.x;
    const int ln = lane & 15;
    const int hb = (lane >> 4) << 3;
    const int t0 = blockIdx.x * 16;
    const int h  = blockIdx.y;
    const size_t baseRow = (size_t)blockIdx.z * TT;

    // ---- S = Q K^T over the span ----
    const _Float16* qrow = q + (baseRow + t0 + ln) * DD + h * DHH;
    v16h aq0 = load_frag(qrow, lane);
    v16h aq1 = load_frag(qrow + 32, lane);

    v8f s[9];
#pragma unroll
    for (int jt = 0; jt < 9; ++jt) {
        const int j = jt * 16 + ln;                 // span column for this lane
        int pos = t0 - WR + j;
        pos = pos < 0 ? 0 : (pos > TT - 1 ? TT - 1 : pos);
        const _Float16* kr = k + (baseRow + pos) * DD + h * DHH;
        v8f a = {};
        a = wmma16(aq0, load_frag(kr, lane), a);
        a = wmma16(aq1, load_frag(kr + 32, lane), a);
        s[jt] = a;
    }

    // ---- mask + scale, row max ----
    float rmax[8];
#pragma unroll
    for (int r = 0; r < 8; ++r) rmax[r] = -1e30f;
#pragma unroll
    for (int jt = 0; jt < 9; ++jt) {
        const int j = jt * 16 + ln;
        const int pos = t0 - WR + j;
#pragma unroll
        for (int r = 0; r < 8; ++r) {
            const int rel = j - (r + hb);
            const bool ok = (rel >= 0) && (rel <= 2 * WR) && (pos >= 0) && (pos < TT);
            const float val = ok ? s[jt][r] * 0.125f : -1e30f;   // 1/sqrt(64)
            s[jt][r] = val;
            rmax[r] = fmaxf(rmax[r], val);
        }
    }
#pragma unroll
    for (int r = 0; r < 8; ++r)
#pragma unroll
        for (int off = 1; off < 16; off <<= 1)
            rmax[r] = fmaxf(rmax[r], __shfl_xor(rmax[r], off, 16));

    // ---- exp + row sum ----
    float rsum[8];
#pragma unroll
    for (int r = 0; r < 8; ++r) rsum[r] = 0.f;
#pragma unroll
    for (int jt = 0; jt < 9; ++jt)
#pragma unroll
        for (int r = 0; r < 8; ++r) {
            const float e = __expf(s[jt][r] - rmax[r]);
            s[jt][r] = e;
            rsum[r] += e;
        }
#pragma unroll
    for (int r = 0; r < 8; ++r) {
#pragma unroll
        for (int off = 1; off < 16; off <<= 1)
            rsum[r] += __shfl_xor(rsum[r], off, 16);
        rsum[r] = 1.f / rsum[r];
    }

    // ---- P -> LDS (rows 0..15, cols 0..143; zero the pad 144..159) ----
#pragma unroll
    for (int jt = 0; jt < 9; ++jt)
#pragma unroll
        for (int r = 0; r < 8; ++r)
            Pb[(r + hb) * SPANP + jt * 16 + ln] = (_Float16)(s[jt][r] * rsum[r]);
    for (int idx = lane; idx < 16 * 16; idx += 32)
        Pb[(idx >> 4) * SPANP + SPAN + (idx & 15)] = (_Float16)0.f;

    // ---- transpose V span into LDS: Vt[d][j] = v[pos(j), h*64+d] ----
    for (int j = lane; j < SPANP; j += 32) {
        int pos = t0 - WR + j;
        pos = pos < 0 ? 0 : (pos > TT - 1 ? TT - 1 : pos);
        const _Float16* vr = v + (baseRow + pos) * DD + h * DHH;
#pragma unroll
        for (int c = 0; c < 8; ++c) {
            v8h chunk = *(const v8h*)(vr + c * 8);
#pragma unroll
            for (int e = 0; e < 8; ++e)
                Vt[(c * 8 + e) * SPANP + j] = chunk[e];
        }
    }
    __syncthreads();

    // ---- O = P Vspan  (M=16, K=160, N=64) ----
    v8f oacc[4] = {};
#pragma unroll
    for (int k0 = 0; k0 < SPANP; k0 += 32) {
        v16h pa = load_frag(&Pb[ln * SPANP + k0], lane);
#pragma unroll
        for (int nt = 0; nt < 4; ++nt) {
            v16h vb = load_frag(&Vt[(nt * 16 + ln) * SPANP + k0], lane);
            oacc[nt] = wmma16(pa, vb, oacc[nt]);
        }
    }
#pragma unroll
    for (int nt = 0; nt < 4; ++nt)
#pragma unroll
        for (int r = 0; r < 8; ++r)
            o[(baseRow + t0 + hb + r) * DD + h * DHH + nt * 16 + ln] =
                (_Float16)oacc[nt][r];
}

// ---------------------------------------------------------------------------
extern "C" void kernel_launch(void* const* d_in, const int* in_sizes, int n_in,
                              void* d_out, int out_size, void* d_ws, size_t ws_size,
                              hipStream_t stream) {
    const float* x  = (const float*)d_in[0];
    const float* wq = (const float*)d_in[1];
    const float* bq = (const float*)d_in[2];
    const float* wk = (const float*)d_in[3];
    const float* bk = (const float*)d_in[4];
    const float* wv = (const float*)d_in[5];
    const float* bv = (const float*)d_in[6];
    const float* wo = (const float*)d_in[7];
    const float* bo = (const float*)d_in[8];
    float* out = (float*)d_out;

    char* ws = (char*)d_ws;
    const size_t MB = 1ull << 20;
    _Float16* x16  = (_Float16*)(ws + 0 * MB);    // 4 MB
    _Float16* wq16 = (_Float16*)(ws + 4 * MB);    // 2 MB
    _Float16* wk16 = (_Float16*)(ws + 6 * MB);    // 2 MB
    _Float16* wv16 = (_Float16*)(ws + 8 * MB);    // 2 MB
    _Float16* wo16 = (_Float16*)(ws + 10 * MB);   // 2 MB
    _Float16* qb   = (_Float16*)(ws + 12 * MB);   // 4 MB
    _Float16* kb   = (_Float16*)(ws + 16 * MB);   // 4 MB
    _Float16* vb   = (_Float16*)(ws + 20 * MB);   // 4 MB
    _Float16* ab   = (_Float16*)(ws + 24 * MB);   // 4 MB  => 28 MB total

    cast_f32_to_f16<<<512, 256, 0, stream>>>(x,  x16,  NBATCH * TT * DD);
    cast_f32_to_f16<<<512, 256, 0, stream>>>(wq, wq16, DD * DD);
    cast_f32_to_f16<<<512, 256, 0, stream>>>(wk, wk16, DD * DD);
    cast_f32_to_f16<<<512, 256, 0, stream>>>(wv, wv16, DD * DD);
    cast_f32_to_f16<<<512, 256, 0, stream>>>(wo, wo16, DD * DD);

    dim3 gQKV(DD / 64, (NBATCH * TT) / 256, 3);
    gemm_qkv<<<gQKV, 256, 0, stream>>>(x16, wq16, wk16, wv16, bq, bk, bv,
                                       qb, kb, vb);

    dim3 gA(TT / 16, HH, NBATCH);
    attn_win<<<gA, 32, 0, stream>>>(qb, kb, vb, ab);

    dim3 gO(DD / 64, (NBATCH * TT) / 256, 1);
    gemm_out<<<gO, 256, 0, stream>>>(ab, wo16, bo, out);
}
